// VQRAEcontrast_46849503265295
// MI455X (gfx1250) — compile-verified, matrix operand order checked by reading, not run on hardware
//
#include <hip/hip_runtime.h>
#include <hip/hip_bf16.h>
#include <cmath>

// ---------------------------------------------------------------------------
// VRAE-style forward on gfx1250: bf16 WMMA GEMMs + QRNN scans + planar flow
// Round 3: K-loop unrolled x2 with ping-pong fragment buffers (no register
// rotation moves), per-lane address math hoisted out of the loop.
// Requires K % 64 == 0 (true for all layers here: 64/512/576/640).
// ---------------------------------------------------------------------------

typedef __bf16 bf16_t;
typedef __attribute__((ext_vector_type(16))) __bf16 v16bf;
typedef __attribute__((ext_vector_type(8)))  __bf16 v8bf;
typedef __attribute__((ext_vector_type(8)))  float  v8f;

constexpr int Tn  = 128;
constexpr int Bn  = 256;
constexpr int DXn = 64;
constexpr int DHn = 512;
constexpr int DZn = 128;
constexpr int Kf  = 16;
constexpr int Mrows = Tn * Bn;            // 32768 rows for every GEMM

constexpr int REV_LD = DXn + DHn;         // 576  (rev_in row width)
constexpr int DEC_LD = DZn + DHn;         // 640  (decoder concat row width)

// d_out offsets (floats), tuple return order
constexpr size_t SZ_BTDX = (size_t)Bn * Tn * DXn;
constexpr size_t SZ_BTDZ = (size_t)Bn * Tn * DZn;
constexpr size_t OFF_DEC_MEAN   = 0;
constexpr size_t OFF_DEC_STD    = OFF_DEC_MEAN + SZ_BTDX;
constexpr size_t OFF_ENC_MEAN   = OFF_DEC_STD + SZ_BTDX;
constexpr size_t OFF_ENC_STD    = OFF_ENC_MEAN + SZ_BTDZ;
constexpr size_t OFF_PRIOR_MEAN = OFF_ENC_STD + SZ_BTDZ;
constexpr size_t OFF_PRIOR_STD  = OFF_PRIOR_MEAN + SZ_BTDZ;
constexpr size_t OFF_ZK         = OFF_PRIOR_STD + SZ_BTDZ;
constexpr size_t OFF_SLDJ       = OFF_ZK + SZ_BTDZ;

#define ACT_NONE     0
#define ACT_RELU     1
#define ACT_SIGMOID  2
#define ACT_SOFTPLUS 3

__device__ __forceinline__ float sigmoidf_(float x) { return 1.0f / (1.0f + __expf(-x)); }
__device__ __forceinline__ float softplusf_(float x) {
    return (x > 20.0f) ? x : log1pf(__expf(x));
}

template <int ACT>
__device__ __forceinline__ float apply_act(float v) {
    if (ACT == ACT_RELU)     return fmaxf(v, 0.0f);
    if (ACT == ACT_SIGMOID)  return sigmoidf_(v);
    if (ACT == ACT_SOFTPLUS) return softplusf_(v);
    return v;
}

// Load one 16x32 bf16 fragment given a per-lane pre-offset pointer.
// ISA 16-bit A layout: lane&15 = row; lanes>=16 take K+8; per lane two
// contiguous 8-element (16 B) chunks at k and k+16.
__device__ __forceinline__ v16bf ldfrag(const bf16_t* __restrict__ p) {
    union { v16bf v; v8bf h[2]; } u;
    u.h[0] = *(const v8bf*)(p);
    u.h[1] = *(const v8bf*)(p + 16);
    return u.v;
}

// ---------------------------------------------------------------------------
// GEMM: C(M,N) = act(A(M,K) @ W(N,K)^T + bias)
// block = 256 threads = 8 waves; wave w owns rows [by*256 + w*32, +32),
// cols [bx*64, +64): 2 A-fragments x 4 B-fragments -> 8 accumulators.
// K-loop unrolled x2 with ping-pong buffers: stage loads write directly
// into the register set the next stage consumes (no rotation moves).
// ---------------------------------------------------------------------------
template <int ACT, bool WF32, bool WBF16>
__global__ __launch_bounds__(256)
void gemm_wmma_bf16(const bf16_t* __restrict__ A, int lda,
                    const bf16_t* __restrict__ Wt, int ldw,
                    const float* __restrict__ bias,
                    float* __restrict__ Cf, int ldcf,
                    bf16_t* __restrict__ Cb, int ldcb,
                    int N, int K) {
    const int wave = threadIdx.x >> 5;
    const int lane = threadIdx.x & 31;
    const int m0 = blockIdx.y * 256 + wave * 32;
    const int n0 = blockIdx.x * 64;

    const int r  = lane & 15;
    const int ko = (lane >> 4) << 3;     // 0 or 8

    // per-lane pre-offset base pointers (only k offset varies in the loop)
    const bf16_t* pa0 = A + (size_t)(m0 + r) * lda + ko;
    const bf16_t* pa1 = A + (size_t)(m0 + 16 + r) * lda + ko;
    const bf16_t* pb0 = Wt + (size_t)(n0 + 0 * 16 + r) * ldw + ko;
    const bf16_t* pb1 = Wt + (size_t)(n0 + 1 * 16 + r) * ldw + ko;
    const bf16_t* pb2 = Wt + (size_t)(n0 + 2 * 16 + r) * ldw + ko;
    const bf16_t* pb3 = Wt + (size_t)(n0 + 3 * 16 + r) * ldw + ko;

    v8f accL[4] = {v8f{}, v8f{}, v8f{}, v8f{}};
    v8f accH[4] = {v8f{}, v8f{}, v8f{}, v8f{}};

    // prologue: fill stage-A buffers with chunk k=0
    v16bf a0A = ldfrag(pa0);
    v16bf a1A = ldfrag(pa1);
    v16bf bA0 = ldfrag(pb0);
    v16bf bA1 = ldfrag(pb1);
    v16bf bA2 = ldfrag(pb2);
    v16bf bA3 = ldfrag(pb3);

    for (int k0 = 0; k0 < K; k0 += 64) {
        // --- stage A computes chunk k0; stage-B loads chunk k0+32 first ---
        const int kb = k0 + 32;                     // always < K (K % 64 == 0)
        __builtin_prefetch(pa0 + k0 + 128, 0, 3);   // far-ahead A stream hint
        v16bf a0B = ldfrag(pa0 + kb);
        v16bf a1B = ldfrag(pa1 + kb);
        v16bf bB0 = ldfrag(pb0 + kb);
        v16bf bB1 = ldfrag(pb1 + kb);
        v16bf bB2 = ldfrag(pb2 + kb);
        v16bf bB3 = ldfrag(pb3 + kb);

        accL[0] = __builtin_amdgcn_wmma_f32_16x16x32_bf16(false, a0A, false, bA0, (short)0, accL[0], false, false);
        accL[1] = __builtin_amdgcn_wmma_f32_16x16x32_bf16(false, a0A, false, bA1, (short)0, accL[1], false, false);
        accL[2] = __builtin_amdgcn_wmma_f32_16x16x32_bf16(false, a0A, false, bA2, (short)0, accL[2], false, false);
        accL[3] = __builtin_amdgcn_wmma_f32_16x16x32_bf16(false, a0A, false, bA3, (short)0, accL[3], false, false);
        accH[0] = __builtin_amdgcn_wmma_f32_16x16x32_bf16(false, a1A, false, bA0, (short)0, accH[0], false, false);
        accH[1] = __builtin_amdgcn_wmma_f32_16x16x32_bf16(false, a1A, false, bA1, (short)0, accH[1], false, false);
        accH[2] = __builtin_amdgcn_wmma_f32_16x16x32_bf16(false, a1A, false, bA2, (short)0, accH[2], false, false);
        accH[3] = __builtin_amdgcn_wmma_f32_16x16x32_bf16(false, a1A, false, bA3, (short)0, accH[3], false, false);

        // --- stage B computes chunk k0+32; stage-A loads chunk k0+64 ------
        const int kn = k0 + 64;
        const int ka = (kn < K) ? kn : 0;           // last reload is harmless dup
        a0A = ldfrag(pa0 + ka);
        a1A = ldfrag(pa1 + ka);
        bA0 = ldfrag(pb0 + ka);
        bA1 = ldfrag(pb1 + ka);
        bA2 = ldfrag(pb2 + ka);
        bA3 = ldfrag(pb3 + ka);

        accL[0] = __builtin_amdgcn_wmma_f32_16x16x32_bf16(false, a0B, false, bB0, (short)0, accL[0], false, false);
        accL[1] = __builtin_amdgcn_wmma_f32_16x16x32_bf16(false, a0B, false, bB1, (short)0, accL[1], false, false);
        accL[2] = __builtin_amdgcn_wmma_f32_16x16x32_bf16(false, a0B, false, bB2, (short)0, accL[2], false, false);
        accL[3] = __builtin_amdgcn_wmma_f32_16x16x32_bf16(false, a0B, false, bB3, (short)0, accL[3], false, false);
        accH[0] = __builtin_amdgcn_wmma_f32_16x16x32_bf16(false, a1B, false, bB0, (short)0, accH[0], false, false);
        accH[1] = __builtin_amdgcn_wmma_f32_16x16x32_bf16(false, a1B, false, bB1, (short)0, accH[1], false, false);
        accH[2] = __builtin_amdgcn_wmma_f32_16x16x32_bf16(false, a1B, false, bB2, (short)0, accH[2], false, false);
        accH[3] = __builtin_amdgcn_wmma_f32_16x16x32_bf16(false, a1B, false, bB3, (short)0, accH[3], false, false);
    }

    const int nl = lane & 15;
    const int mo = (lane >> 4) ? 8 : 0;
#pragma unroll
    for (int j = 0; j < 4; ++j) {
        const int n = n0 + j * 16 + nl;
        if (n >= N) continue;
        const float bv = bias[n];
#pragma unroll
        for (int rr = 0; rr < 8; ++rr) {
            const int mL = m0 + mo + rr;
            float vL = apply_act<ACT>(accL[j][rr] + bv);
            if (WF32)  Cf[(size_t)mL * ldcf + n] = vL;
            if (WBF16) Cb[(size_t)mL * ldcb + n] = (bf16_t)vL;
            const int mH = m0 + 16 + mo + rr;
            float vH = apply_act<ACT>(accH[j][rr] + bv);
            if (WF32)  Cf[(size_t)mH * ldcf + n] = vH;
            if (WBF16) Cb[(size_t)mH * ldcb + n] = (bf16_t)vH;
        }
    }
}

// ---------------------------------------------------------------------------
// Elementwise converts
// ---------------------------------------------------------------------------
__global__ void cvt_f32_to_bf16(const float* __restrict__ in,
                                bf16_t* __restrict__ out, int n) {
    int i = blockIdx.x * blockDim.x + threadIdx.x;
    if (i < n) out[i] = (bf16_t)in[i];
}

// rev_in[t][b][0:64] = y[T-1-t][b][:]
__global__ void pack_y_rev(const float* __restrict__ y,
                           bf16_t* __restrict__ rev_in) {
    int idx = blockIdx.x * blockDim.x + threadIdx.x;
    if (idx >= Tn * Bn * DXn) return;
    int j  = idx % DXn;
    int tb = idx / DXn;
    int b  = tb % Bn;
    int t  = tb / Bn;
    rev_in[((size_t)(Tn - 1 - t) * Bn + b) * REV_LD + j] = (bf16_t)y[idx];
}

// ---------------------------------------------------------------------------
// QRNN gate + forget-mult scan; one thread per (b,h), sequential over T.
// MODE 1: h of hidden QRNN -> rev_in[(T-1-t)][b][64+h], dec_in[b][t][128+h]
// MODE 2: a of inference QRNN (already reversed time) -> rev_a in (B,T,DH)
// ---------------------------------------------------------------------------
template <int MODE>
__global__ __launch_bounds__(256)
void qrnn_scan(const float* __restrict__ Y,
               bf16_t* __restrict__ outA,
               bf16_t* __restrict__ outB) {
    const int tid = blockIdx.x * blockDim.x + threadIdx.x;
    const int b = tid / DHn;
    const int h = tid % DHn;
    float c = 0.0f;
    for (int t = 0; t < Tn; ++t) {
        const size_t base = ((size_t)t * Bn + b) * (3 * DHn);
        const float z = tanhf(Y[base + h]);
        const float f = sigmoidf_(Y[base + DHn + h]);
        const float o = sigmoidf_(Y[base + 2 * DHn + h]);
        c = f * z + (1.0f - f) * c;
        const float out = o * c;
        if (MODE == 1) {
            outA[((size_t)(Tn - 1 - t) * Bn + b) * REV_LD + DXn + h] = (bf16_t)out;
            outB[((size_t)b * Tn + t) * DEC_LD + DZn + h] = (bf16_t)out;
        } else {
            outA[((size_t)b * Tn + (Tn - 1 - t)) * DHn + h] = (bf16_t)out;
        }
    }
}

// ---------------------------------------------------------------------------
// Planar flow precompute: u_hat[k][:], wu[k] = dot(w_k, u_hat_k).
// One wave per k (block = 512 threads = 16 waves).
// ---------------------------------------------------------------------------
__global__ __launch_bounds__(512)
void flow_pre(const float* __restrict__ u, const float* __restrict__ w,
              float* __restrict__ u_hat, float* __restrict__ wu) {
    const int k = threadIdx.x >> 5;
    const int lane = threadIdx.x & 31;
    float wv[4], uv[4];
    float udw = 0.0f, n2 = 0.0f;
#pragma unroll
    for (int i = 0; i < 4; ++i) {
        const int j = lane + 32 * i;
        wv[i] = w[k * DZn + j];
        uv[i] = u[k * DZn + j];
        udw += uv[i] * wv[i];
        n2  += wv[i] * wv[i];
    }
#pragma unroll
    for (int off = 16; off >= 1; off >>= 1) {
        udw += __shfl_xor(udw, off, 32);
        n2  += __shfl_xor(n2, off, 32);
    }
    const float inv  = rsqrtf(n2);
    const float mm   = softplusf_(udw) - 1.0f;
    const float coef = mm - udw;
    float wupart = 0.0f;
#pragma unroll
    for (int i = 0; i < 4; ++i) {
        const int j = lane + 32 * i;
        const float uh = coef * wv[i] * inv + uv[i];
        u_hat[k * DZn + j] = uh;
        wupart += wv[i] * uh;
    }
#pragma unroll
    for (int off = 16; off >= 1; off >>= 1) wupart += __shfl_xor(wupart, off, 32);
    if (lane == 0) wu[k] = wupart;
}

// ---------------------------------------------------------------------------
// z0 = enc_mean + enc_std * eps, then K planar-flow steps.
// One wave per (b,t) row; DZ=128 -> 4 values per lane.
// ---------------------------------------------------------------------------
__global__ __launch_bounds__(256)
void planar_flow(const float* __restrict__ enc_mean,
                 const float* __restrict__ enc_std,
                 const float* __restrict__ eps,
                 const float* __restrict__ w,
                 const float* __restrict__ bflow,
                 const float* __restrict__ u_hat,
                 const float* __restrict__ wu,
                 float* __restrict__ zk_out,
                 bf16_t* __restrict__ dec_in,
                 float* __restrict__ sldj_out) {
    const int row  = blockIdx.x * 8 + (threadIdx.x >> 5);
    const int lane = threadIdx.x & 31;
    float z[4];
#pragma unroll
    for (int i = 0; i < 4; ++i) {
        const size_t idx = (size_t)row * DZn + lane + 32 * i;
        z[i] = enc_mean[idx] + enc_std[idx] * eps[idx];
    }
    float sldj = 0.0f;
    for (int k = 0; k < Kf; ++k) {
        float d = 0.0f;
#pragma unroll
        for (int i = 0; i < 4; ++i) d += z[i] * w[k * DZn + lane + 32 * i];
#pragma unroll
        for (int off = 16; off >= 1; off >>= 1) d += __shfl_xor(d, off, 32);
        const float tt = tanhf(d + bflow[k]);
#pragma unroll
        for (int i = 0; i < 4; ++i) z[i] += u_hat[k * DZn + lane + 32 * i] * tt;
        const float psi = (1.0f - tt * tt) * wu[k];
        sldj -= __logf(fabsf(psi + 1.0f) + 1e-8f);
    }
#pragma unroll
    for (int i = 0; i < 4; ++i) {
        const int j = lane + 32 * i;
        const size_t idx = (size_t)row * DZn + j;
        zk_out[idx] = z[i];
        dec_in[(size_t)row * DEC_LD + j] = (bf16_t)z[i];
    }
    if (lane == 0) sldj_out[row] = sldj;
}

// ---------------------------------------------------------------------------
// Launcher
// ---------------------------------------------------------------------------
extern "C" void kernel_launch(void* const* d_in, const int* in_sizes, int n_in,
                              void* d_out, int out_size, void* d_ws, size_t ws_size,
                              hipStream_t stream) {
    const float* x        = (const float*)d_in[0];
    const float* y        = (const float*)d_in[1];
    const float* eps      = (const float*)d_in[2];
    const float* hs_W     = (const float*)d_in[3];
    const float* hs_b     = (const float*)d_in[4];
    const float* q_W      = (const float*)d_in[5];
    const float* q_b      = (const float*)d_in[6];
    const float* enc_phi_W  = (const float*)d_in[7];
    const float* enc_phi_b  = (const float*)d_in[8];
    const float* enc_mean_W = (const float*)d_in[9];
    const float* enc_mean_b = (const float*)d_in[10];
    const float* enc_std_W  = (const float*)d_in[11];
    const float* enc_std_b  = (const float*)d_in[12];
    const float* prior_phi_W  = (const float*)d_in[13];
    const float* prior_phi_b  = (const float*)d_in[14];
    const float* prior_mean_W = (const float*)d_in[15];
    const float* prior_mean_b = (const float*)d_in[16];
    const float* prior_std_W  = (const float*)d_in[17];
    const float* prior_std_b  = (const float*)d_in[18];
    const float* dec_phi_W1 = (const float*)d_in[19];
    const float* dec_phi_b1 = (const float*)d_in[20];
    const float* dec_phi_W2 = (const float*)d_in[21];
    const float* dec_phi_b2 = (const float*)d_in[22];
    const float* dec_mean_W = (const float*)d_in[23];
    const float* dec_mean_b = (const float*)d_in[24];
    const float* dec_std_W  = (const float*)d_in[25];
    const float* dec_std_b  = (const float*)d_in[26];
    const float* pnf_u = (const float*)d_in[27];
    const float* pnf_w = (const float*)d_in[28];
    const float* pnf_b = (const float*)d_in[29];

    float* out = (float*)d_out;

    // ---- carve workspace (256B aligned) ----
    size_t off = 0;
    auto carve = [&](size_t bytes) -> void* {
        off = (off + 255) & ~(size_t)255;
        void* p = (char*)d_ws + off;
        off += bytes;
        return p;
    };
    bf16_t* xb       = (bf16_t*)carve((size_t)Mrows * DXn * 2);
    bf16_t* rev_in   = (bf16_t*)carve((size_t)Mrows * REV_LD * 2);
    bf16_t* dec_in   = (bf16_t*)carve((size_t)Mrows * DEC_LD * 2);
    bf16_t* enc_in   = (bf16_t*)carve((size_t)Mrows * DHn * 2);
    bf16_t* e1       = (bf16_t*)carve((size_t)Mrows * DHn * 2);
    bf16_t* e2       = (bf16_t*)carve((size_t)Mrows * DHn * 2);
    float*  Ybuf     = (float*)carve((size_t)Mrows * 3 * DHn * 4);
    float*  u_hat    = (float*)carve((size_t)Kf * DZn * 4);
    float*  wu       = (float*)carve((size_t)Kf * 4);
    bf16_t* hsWb     = (bf16_t*)carve((size_t)3 * DHn * DXn * 2);
    bf16_t* qWb      = (bf16_t*)carve((size_t)3 * DHn * REV_LD * 2);
    bf16_t* encPhiWb = (bf16_t*)carve((size_t)2 * DHn * DHn * 2);
    bf16_t* encMeanWb= (bf16_t*)carve((size_t)DZn * DHn * 2);
    bf16_t* encStdWb = (bf16_t*)carve((size_t)DZn * DHn * 2);
    bf16_t* priPhiWb = (bf16_t*)carve((size_t)2 * DHn * DHn * 2);
    bf16_t* priMeanWb= (bf16_t*)carve((size_t)DZn * DHn * 2);
    bf16_t* priStdWb = (bf16_t*)carve((size_t)DZn * DHn * 2);
    bf16_t* decW1b   = (bf16_t*)carve((size_t)DHn * DEC_LD * 2);
    bf16_t* decW2b   = (bf16_t*)carve((size_t)DHn * DHn * 2);
    bf16_t* decMeanWb= (bf16_t*)carve((size_t)DXn * DHn * 2);
    bf16_t* decStdWb = (bf16_t*)carve((size_t)DXn * DHn * 2);

    auto cvt = [&](const float* src, bf16_t* dst, size_t n) {
        cvt_f32_to_bf16<<<(unsigned)((n + 255) / 256), 256, 0, stream>>>(src, dst, (int)n);
    };

    // ---- operand conversion ----
    cvt(x, xb, (size_t)Mrows * DXn);
    pack_y_rev<<<(Tn * Bn * DXn + 255) / 256, 256, 0, stream>>>(y, rev_in);
    cvt(hs_W, hsWb, (size_t)3 * DHn * DXn);
    cvt(q_W, qWb, (size_t)3 * DHn * REV_LD);
    cvt(enc_phi_W, encPhiWb, (size_t)2 * DHn * DHn);
    cvt(enc_mean_W, encMeanWb, (size_t)DZn * DHn);
    cvt(enc_std_W, encStdWb, (size_t)DZn * DHn);
    cvt(prior_phi_W, priPhiWb, (size_t)2 * DHn * DHn);
    cvt(prior_mean_W, priMeanWb, (size_t)DZn * DHn);
    cvt(prior_std_W, priStdWb, (size_t)DZn * DHn);
    cvt(dec_phi_W1, decW1b, (size_t)DHn * DEC_LD);
    cvt(dec_phi_W2, decW2b, (size_t)DHn * DHn);
    cvt(dec_mean_W, decMeanWb, (size_t)DXn * DHn);
    cvt(dec_std_W, decStdWb, (size_t)DXn * DHn);

    const dim3 blk(256);
    auto grid_for = [](int N) { return dim3((unsigned)((N + 63) / 64),
                                            (unsigned)(Mrows / 256)); };

    // ---- hidden QRNN ----
    gemm_wmma_bf16<ACT_NONE, true, false><<<grid_for(3 * DHn), blk, 0, stream>>>(
        xb, DXn, hsWb, DXn, hs_b, Ybuf, 3 * DHn, nullptr, 0, 3 * DHn, DXn);
    qrnn_scan<1><<<(Bn * DHn) / 256, blk, 0, stream>>>(Ybuf, rev_in, dec_in);

    // ---- inference QRNN over reversed [y, h] ----
    gemm_wmma_bf16<ACT_NONE, true, false><<<grid_for(3 * DHn), blk, 0, stream>>>(
        rev_in, REV_LD, qWb, REV_LD, q_b, Ybuf, 3 * DHn, nullptr, 0, 3 * DHn, REV_LD);
    qrnn_scan<2><<<(Bn * DHn) / 256, blk, 0, stream>>>(Ybuf, enc_in, nullptr);

    // ---- encoder ----
    gemm_wmma_bf16<ACT_RELU, false, true><<<grid_for(DHn), blk, 0, stream>>>(
        enc_in, DHn, encPhiWb, DHn, enc_phi_b, nullptr, 0, e1, DHn, DHn, DHn);
    gemm_wmma_bf16<ACT_RELU, false, true><<<grid_for(DHn), blk, 0, stream>>>(
        e1, DHn, encPhiWb + (size_t)DHn * DHn, DHn, enc_phi_b + DHn,
        nullptr, 0, e2, DHn, DHn, DHn);
    gemm_wmma_bf16<ACT_SIGMOID, true, false><<<grid_for(DZn), blk, 0, stream>>>(
        e2, DHn, encMeanWb, DHn, enc_mean_b, out + OFF_ENC_MEAN, DZn, nullptr, 0, DZn, DHn);
    gemm_wmma_bf16<ACT_SOFTPLUS, true, false><<<grid_for(DZn), blk, 0, stream>>>(
        e2, DHn, encStdWb, DHn, enc_std_b, out + OFF_ENC_STD, DZn, nullptr, 0, DZn, DHn);

    // ---- planar flow ----
    flow_pre<<<1, 512, 0, stream>>>(pnf_u, pnf_w, u_hat, wu);
    planar_flow<<<Mrows / 8, blk, 0, stream>>>(
        out + OFF_ENC_MEAN, out + OFF_ENC_STD, eps, pnf_w, pnf_b, u_hat, wu,
        out + OFF_ZK, dec_in, out + OFF_SLDJ);

    // ---- prior (A = h part of dec_in, lda = 640) ----
    gemm_wmma_bf16<ACT_RELU, false, true><<<grid_for(DHn), blk, 0, stream>>>(
        dec_in + DZn, DEC_LD, priPhiWb, DHn, prior_phi_b, nullptr, 0, e1, DHn, DHn, DHn);
    gemm_wmma_bf16<ACT_RELU, false, true><<<grid_for(DHn), blk, 0, stream>>>(
        e1, DHn, priPhiWb + (size_t)DHn * DHn, DHn, prior_phi_b + DHn,
        nullptr, 0, e2, DHn, DHn, DHn);
    gemm_wmma_bf16<ACT_SIGMOID, true, false><<<grid_for(DZn), blk, 0, stream>>>(
        e2, DHn, priMeanWb, DHn, prior_mean_b, out + OFF_PRIOR_MEAN, DZn, nullptr, 0, DZn, DHn);
    gemm_wmma_bf16<ACT_SOFTPLUS, true, false><<<grid_for(DZn), blk, 0, stream>>>(
        e2, DHn, priStdWb, DHn, prior_std_b, out + OFF_PRIOR_STD, DZn, nullptr, 0, DZn, DHn);

    // ---- decoder ----
    gemm_wmma_bf16<ACT_RELU, false, true><<<grid_for(DHn), blk, 0, stream>>>(
        dec_in, DEC_LD, decW1b, DEC_LD, dec_phi_b1, nullptr, 0, e1, DHn, DHn, DEC_LD);
    gemm_wmma_bf16<ACT_RELU, false, true><<<grid_for(DHn), blk, 0, stream>>>(
        e1, DHn, decW2b, DHn, dec_phi_b2, nullptr, 0, e2, DHn, DHn, DHn);
    gemm_wmma_bf16<ACT_SIGMOID, true, false><<<grid_for(DXn), blk, 0, stream>>>(
        e2, DHn, decMeanWb, DHn, dec_mean_b, out + OFF_DEC_MEAN, DXn, nullptr, 0, DXn, DHn);
    gemm_wmma_bf16<ACT_SOFTPLUS, true, false><<<grid_for(DXn), blk, 0, stream>>>(
        e2, DHn, decStdWb, DHn, dec_std_b, out + OFF_DEC_STD, DXn, nullptr, 0, DXn, DHn);

    (void)in_sizes; (void)n_in; (void)out_size; (void)ws_size;
}
